// AttentionBlock_1726576853730
// MI455X (gfx1250) — matrix operand order repaired and checked by reading, not compiled, not run on hardware
//
#include <hip/hip_runtime.h>

// ---------------------------------------------------------------------------
// AttentionBlock for MI455X (gfx1250): bf16 WMMA flash-attention pipeline.
// B=2, S=2048, D=1024, H=16, HD=64.
// GEMMs use async global->LDS (ASYNCcnt) double-buffered tiles; weights are
// pre-transposed once so every tile copy is a contiguous b128 async load.
// ---------------------------------------------------------------------------

#define BB 2
#define SS 2048
#define DD 1024
#define HH 16
#define HDIM 64
#define MROWS (BB * SS)   // 4096
#define N3 (3 * DD)       // 3072

typedef __attribute__((ext_vector_type(16))) __bf16 v16bf;
typedef __attribute__((ext_vector_type(8)))  float  v8f;
typedef __attribute__((ext_vector_type(4)))  unsigned int v4u;

union FragBF {
    v16bf v;
    unsigned int u[8];
    v4u q[2];
};

__device__ __forceinline__ unsigned short f2bf(float f) {
    unsigned int u = __float_as_uint(f);
    unsigned int r = (u + 0x7FFFu + ((u >> 16) & 1u)) >> 16;
    return (unsigned short)r;
}

#define WMMA_BF16(A, Bm, C) \
    __builtin_amdgcn_wmma_f32_16x16x32_bf16(false, (A), false, (Bm), (short)0, (C), false, false)

// Async global -> LDS 128-bit copy (CDNA5 GLOBAL_LOAD_ASYNC_TO_LDS_B128,
// tracked by ASYNCcnt). VDST = per-lane LDS byte address (generic pointer
// truncated to 32 bits per the ISA aperture rules), VADDR = 64-bit global.
__device__ __forceinline__ void async_copy_b128(const void* gptr, void* lptr) {
    unsigned int lds  = (unsigned int)(size_t)lptr;
    unsigned long long g = (unsigned long long)(size_t)gptr;
    asm volatile("global_load_async_to_lds_b128 %0, %1, off"
                 :: "v"(lds), "v"(g) : "memory");
}

__device__ __forceinline__ void wait_async0() {
    asm volatile("s_wait_asynccnt 0x0" ::: "memory");
}

// ---------------------------------------------------------------------------
// Kernel 1a: fp32 -> bf16 (round to nearest even)
// ---------------------------------------------------------------------------
__global__ void f32_to_bf16_kernel(const float* __restrict__ in,
                                   unsigned short* __restrict__ out, int n) {
    int i = blockIdx.x * blockDim.x + threadIdx.x;
    int stride = gridDim.x * blockDim.x;
    for (; i < n; i += stride) out[i] = f2bf(in[i]);
}

// ---------------------------------------------------------------------------
// Kernel 1b: fp32 [K,N] -> bf16 transposed [N,K]   (one-time weight prep)
// ---------------------------------------------------------------------------
__global__ void f32_to_bf16_T_kernel(const float* __restrict__ in,
                                     unsigned short* __restrict__ out,
                                     int K, int N) {
    int i = blockIdx.x * blockDim.x + threadIdx.x;
    int stride = gridDim.x * blockDim.x;
    int total = K * N;
    for (; i < total; i += stride) {
        int k = i / N, n = i - k * N;           // contiguous reads over n
        out[(size_t)n * K + k] = f2bf(in[i]);
    }
}

// ---------------------------------------------------------------------------
// Shared GEMM tile machinery.
// Tile: 128(M) x 64(N), K-step 32. 8 waves: 4 along M x 2 along N,
// each wave owns a 32x32 subtile = 2x2 WMMA fragments.
// LDS per buffer: sA [128][32] row-major, sB [64][32] N-major (from W^T).
// Double-buffered, filled with async b128 copies.
// ---------------------------------------------------------------------------
#define TILE_A (128 * 32)
#define TILE_B (64 * 32)
#define TILE_AB (TILE_A + TILE_B)

__device__ __forceinline__ void issue_tile_async(
    const unsigned short* __restrict__ A,    // [*,1024] row-major
    const unsigned short* __restrict__ WT,   // [N][1024] (pre-transposed W)
    unsigned short* sA, unsigned short* sB,
    int m0, int n0, int k0, int tid) {
    int r = tid >> 2, c = tid & 3;           // 4 x 16B chunks per 32-half row
    async_copy_b128(A  + (size_t)(m0 + r)      * DD + k0 + c * 8, sA + r * 32 + c * 8);
    async_copy_b128(A  + (size_t)(m0 + r + 64) * DD + k0 + c * 8, sA + (r + 64) * 32 + c * 8);
    async_copy_b128(WT + (size_t)(n0 + r)      * DD + k0 + c * 8, sB + r * 32 + c * 8);
}

__device__ __forceinline__ void gemm_tile_accum(
    const unsigned short* sA, const unsigned short* sB,
    int waveM, int waveN, int g, int lm, v8f acc[2][2]) {
    FragBF aF[2];
#pragma unroll
    for (int sub = 0; sub < 2; ++sub) {
        int row = waveM * 32 + sub * 16 + lm;
#pragma unroll
        for (int p = 0; p < 8; ++p) {
            int k2 = (p << 1) + (g << 4);
            aF[sub].u[p] = *(const unsigned int*)(sA + row * 32 + k2);
        }
    }
    FragBF bF[2];
#pragma unroll
    for (int nt = 0; nt < 2; ++nt) {
        int col = waveN * 32 + nt * 16 + lm;
        const unsigned short* base = sB + col * 32 + g * 16;
        bF[nt].q[0] = *(const v4u*)(base);
        bF[nt].q[1] = *(const v4u*)(base + 8);
    }
#pragma unroll
    for (int sub = 0; sub < 2; ++sub)
#pragma unroll
        for (int nt = 0; nt < 2; ++nt)
            acc[sub][nt] = WMMA_BF16(aF[sub].v, bF[nt].v, acc[sub][nt]);
}

// ---------------------------------------------------------------------------
// Kernel 2: QKV projection GEMM  [4096,1024] x W^T[3072,1024] + bias.
// Scatters Q (scaled 1/8), K row-major, V transposed for the attention stage.
// ---------------------------------------------------------------------------
__global__ __launch_bounds__(256) void qkv_gemm_kernel(
    const unsigned short* __restrict__ A,    // xb   [4096,1024] bf16
    const unsigned short* __restrict__ WT,   // wqkvT [3072,1024] bf16
    const float* __restrict__ bias,          // [3072]
    unsigned short* __restrict__ Qb,         // [32,2048,64]
    unsigned short* __restrict__ Kb,         // [32,2048,64]
    unsigned short* __restrict__ VTb) {      // [32,64,2048]
    __shared__ unsigned short smem[2 * TILE_AB];

    const int tid  = threadIdx.x;
    const int wid  = tid >> 5;
    const int lane = tid & 31;
    const int g    = lane >> 4;
    const int lm   = lane & 15;
    const int m0   = blockIdx.y * 128;
    const int n0   = blockIdx.x * 64;
    const int waveM = wid >> 1;
    const int waveN = wid & 1;

    v8f acc[2][2];
    acc[0][0] = {}; acc[0][1] = {}; acc[1][0] = {}; acc[1][1] = {};

    issue_tile_async(A, WT, smem, smem + TILE_A, m0, n0, 0, tid);
    for (int kt = 0; kt < DD / 32; ++kt) {
        wait_async0();
        __syncthreads();
        int buf = kt & 1;
        if (kt + 1 < DD / 32) {
            int nb = buf ^ 1;
            issue_tile_async(A, WT, smem + nb * TILE_AB,
                             smem + nb * TILE_AB + TILE_A,
                             m0, n0, (kt + 1) * 32, tid);
        }
        gemm_tile_accum(smem + buf * TILE_AB, smem + buf * TILE_AB + TILE_A,
                        waveM, waveN, g, lm, acc);
    }

    // Epilogue: scatter into Q / K / V^T per qkv.reshape(B,S,3,H,HD)
#pragma unroll
    for (int sub = 0; sub < 2; ++sub) {
#pragma unroll
        for (int nt = 0; nt < 2; ++nt) {
            int col   = n0 + waveN * 32 + nt * 16 + lm;
            float bv  = bias[col];
            int three = col >> 10;
            int h     = (col >> 6) & 15;
            int hd    = col & 63;
#pragma unroll
            for (int r = 0; r < 8; ++r) {
                int row = m0 + waveM * 32 + sub * 16 + g * 8 + r;
                int b = row >> 11;
                int s = row & 2047;
                float val = acc[sub][nt][r] + bv;
                int bh = b * HH + h;
                if (three == 0) {
                    Qb[((size_t)bh * SS + s) * HDIM + hd] = f2bf(val * 0.125f);
                } else if (three == 1) {
                    Kb[((size_t)bh * SS + s) * HDIM + hd] = f2bf(val);
                } else {
                    VTb[((size_t)bh * HDIM + hd) * SS + s] = f2bf(val);
                }
            }
        }
    }
}

// ---------------------------------------------------------------------------
// Kernel 3: flash attention. Grid (S/64, B*H), 4 waves/block; each wave owns
// 16 query rows and streams 2048 keys in 32-key tiles. fp32 running max/sum,
// P staged per-wave in LDS (same-wave LDS ordering via s_wait_dscnt).
// ---------------------------------------------------------------------------
__global__ __launch_bounds__(128) void attn_kernel(
    const unsigned short* __restrict__ Qb,
    const unsigned short* __restrict__ Kb,
    const unsigned short* __restrict__ VTb,
    unsigned short* __restrict__ attnb) {    // [4096,1024] bf16
    __shared__ unsigned short pP[4 * 16 * 32];

    const int tid  = threadIdx.x;
    const int wid  = tid >> 5;
    const int lane = tid & 31;
    const int g    = lane >> 4;
    const int lm   = lane & 15;
    const int bh   = blockIdx.y;
    const int q0   = blockIdx.x * 64 + wid * 16;

    const unsigned short* Qp = Qb + (size_t)bh * SS * HDIM;
    const unsigned short* Kp = Kb + (size_t)bh * SS * HDIM;
    const unsigned short* Vt = VTb + (size_t)bh * HDIM * SS;
    unsigned short* pW = pP + wid * 16 * 32;

    FragBF aq[2];
#pragma unroll
    for (int kh = 0; kh < 2; ++kh) {
#pragma unroll
        for (int p = 0; p < 8; ++p) {
            int k2 = (p << 1) + (g << 4) + kh * 32;
            aq[kh].u[p] = *(const unsigned int*)(Qp + (q0 + lm) * HDIM + k2);
        }
    }

    float mrow[8], lrow[8];
    v8f o[4];
#pragma unroll
    for (int r = 0; r < 8; ++r) { mrow[r] = -3.0e38f; lrow[r] = 0.0f; }
    o[0] = {}; o[1] = {}; o[2] = {}; o[3] = {};

    for (int kt = 0; kt < SS / 32; ++kt) {
        v8f c[2];
#pragma unroll
        for (int nt = 0; nt < 2; ++nt) {
            int key = kt * 32 + nt * 16 + lm;
            const unsigned short* kr = Kp + key * HDIM + g * 16;
            FragBF bk0, bk1;
            bk0.q[0] = *(const v4u*)(kr);
            bk0.q[1] = *(const v4u*)(kr + 8);
            bk1.q[0] = *(const v4u*)(kr + 32);
            bk1.q[1] = *(const v4u*)(kr + 40);
            v8f z = {};
            z = WMMA_BF16(aq[0].v, bk0.v, z);
            c[nt] = WMMA_BF16(aq[1].v, bk1.v, z);
        }
        float scale[8];
#pragma unroll
        for (int r = 0; r < 8; ++r) {
            float v = fmaxf(c[0][r], c[1][r]);
            v = fmaxf(v, __shfl_xor(v, 1, 32));
            v = fmaxf(v, __shfl_xor(v, 2, 32));
            v = fmaxf(v, __shfl_xor(v, 4, 32));
            v = fmaxf(v, __shfl_xor(v, 8, 32));
            float nm = fmaxf(mrow[r], v);
            float sc = __expf(mrow[r] - nm);
            float p0 = __expf(c[0][r] - nm);
            float p1 = __expf(c[1][r] - nm);
            float sum = p0 + p1;
            sum += __shfl_xor(sum, 1, 32);
            sum += __shfl_xor(sum, 2, 32);
            sum += __shfl_xor(sum, 4, 32);
            sum += __shfl_xor(sum, 8, 32);
            lrow[r] = lrow[r] * sc + sum;
            mrow[r] = nm;
            scale[r] = sc;
            int row = g * 8 + r;
            pW[row * 32 + lm]      = f2bf(p0);
            pW[row * 32 + lm + 16] = f2bf(p1);
        }
#pragma unroll
        for (int f = 0; f < 4; ++f)
#pragma unroll
            for (int r = 0; r < 8; ++r) o[f][r] *= scale[r];

        asm volatile("s_wait_dscnt 0" ::: "memory");

        FragBF ap;
#pragma unroll
        for (int p = 0; p < 8; ++p) {
            int k2 = (p << 1) + (g << 4);
            ap.u[p] = *(const unsigned int*)(pW + lm * 32 + k2);
        }
#pragma unroll
        for (int f = 0; f < 4; ++f) {
            int hd = f * 16 + lm;
            const unsigned short* vr = Vt + hd * SS + kt * 32 + g * 16;
            FragBF bv;
            bv.q[0] = *(const v4u*)(vr);
            bv.q[1] = *(const v4u*)(vr + 8);
            o[f] = WMMA_BF16(ap.v, bv.v, o[f]);
        }
    }

    int b = bh >> 4, h = bh & 15;
#pragma unroll
    for (int r = 0; r < 8; ++r) {
        float inv = 1.0f / lrow[r];
        int s = q0 + g * 8 + r;
#pragma unroll
        for (int f = 0; f < 4; ++f) {
            int hd = f * 16 + lm;
            attnb[((size_t)(b * SS + s)) * DD + h * HDIM + hd] = f2bf(o[f][r] * inv);
        }
    }
}

// ---------------------------------------------------------------------------
// Kernel 4: output projection GEMM [4096,1024] x W^T[1024,1024] + bias -> fp32
// ---------------------------------------------------------------------------
__global__ __launch_bounds__(256) void out_gemm_kernel(
    const unsigned short* __restrict__ A,    // attnb [4096,1024]
    const unsigned short* __restrict__ WT,   // woutT [1024,1024]
    const float* __restrict__ bias,          // [1024]
    float* __restrict__ out) {               // [4096,1024]
    __shared__ unsigned short smem[2 * TILE_AB];

    const int tid  = threadIdx.x;
    const int wid  = tid >> 5;
    const int lane = tid & 31;
    const int g    = lane >> 4;
    const int lm   = lane & 15;
    const int m0   = blockIdx.y * 128;
    const int n0   = blockIdx.x * 64;
    const int waveM = wid >> 1;
    const int waveN = wid & 1;

    v8f acc[2][2];
    acc[0][0] = {}; acc[0][1] = {}; acc[1][0] = {}; acc[1][1] = {};

    issue_tile_async(A, WT, smem, smem + TILE_A, m0, n0, 0, tid);
    for (int kt = 0; kt < DD / 32; ++kt) {
        wait_async0();
        __syncthreads();
        int buf = kt & 1;
        if (kt + 1 < DD / 32) {
            int nb = buf ^ 1;
            issue_tile_async(A, WT, smem + nb * TILE_AB,
                             smem + nb * TILE_AB + TILE_A,
                             m0, n0, (kt + 1) * 32, tid);
        }
        gemm_tile_accum(smem + buf * TILE_AB, smem + buf * TILE_AB + TILE_A,
                        waveM, waveN, g, lm, acc);
    }

#pragma unroll
    for (int sub = 0; sub < 2; ++sub) {
#pragma unroll
        for (int nt = 0; nt < 2; ++nt) {
            int col  = n0 + waveN * 32 + nt * 16 + lm;
            float bv = bias[col];
#pragma unroll
            for (int r = 0; r < 8; ++r) {
                int row = m0 + waveM * 32 + sub * 16 + g * 8 + r;
                out[(size_t)row * DD + col] = acc[sub][nt][r] + bv;
            }
        }
    }
}

// ---------------------------------------------------------------------------
// Launch
// ---------------------------------------------------------------------------
extern "C" void kernel_launch(void* const* d_in, const int* in_sizes, int n_in,
                              void* d_out, int out_size, void* d_ws, size_t ws_size,
                              hipStream_t stream) {
    const float* x     = (const float*)d_in[0];   // [B,S,D]
    const float* w_qkv = (const float*)d_in[1];   // [D,3D]
    const float* b_qkv = (const float*)d_in[2];   // [3D]
    const float* w_out = (const float*)d_in[3];   // [D,D]
    const float* b_out = (const float*)d_in[4];   // [D]
    float* out = (float*)d_out;

    unsigned short* xb    = (unsigned short*)d_ws;
    unsigned short* wqkvT = xb    + (size_t)MROWS * DD;   // [3072,1024]
    unsigned short* woutT = wqkvT + (size_t)DD * N3;      // [1024,1024]
    unsigned short* Qb    = woutT + (size_t)DD * DD;
    unsigned short* Kb    = Qb    + (size_t)BB * HH * SS * HDIM;
    unsigned short* VTb   = Kb    + (size_t)BB * HH * SS * HDIM;
    unsigned short* attnb = VTb   + (size_t)BB * HH * HDIM * SS;

    f32_to_bf16_kernel<<<512, 256, 0, stream>>>(x, xb, MROWS * DD);
    f32_to_bf16_T_kernel<<<512, 256, 0, stream>>>(w_qkv, wqkvT, DD, N3);
    f32_to_bf16_T_kernel<<<512, 256, 0, stream>>>(w_out, woutT, DD, DD);

    qkv_gemm_kernel<<<dim3(N3 / 64, MROWS / 128), 256, 0, stream>>>(
        xb, wqkvT, b_qkv, Qb, Kb, VTb);

    attn_kernel<<<dim3(SS / 64, BB * HH), 128, 0, stream>>>(Qb, Kb, VTb, attnb);

    out_gemm_kernel<<<dim3(DD / 64, MROWS / 128), 256, 0, stream>>>(
        attnb, woutT, b_out, out);
}
